// MultiheadSelfAttention_22634477650226
// MI455X (gfx1250) — matrix operand order repaired and checked by reading, not compiled, not run on hardware
//
#include <hip/hip_runtime.h>
#include <hip/hip_bf16.h>

// Problem constants (reference: B=2, T=4096, D=768, H=12, HD=64)
#define BSZ   2
#define TLEN  4096
#define DMOD  768
#define NH    12
#define HDIM  64
#define NROW  (BSZ * TLEN)     // 8192
#define QKVN  (3 * DMOD)       // 2304

typedef __attribute__((ext_vector_type(16))) __bf16 v16bf;
typedef __attribute__((ext_vector_type(8)))  float  v8f;
typedef int v4i __attribute__((vector_size(16)));   // matches builtin's param type

#define AS1 __attribute__((address_space(1)))
#define AS3 __attribute__((address_space(3)))

#if defined(__AMDGCN__) && \
    __has_builtin(__builtin_amdgcn_global_load_async_to_lds_b128) && \
    __has_builtin(__builtin_amdgcn_s_wait_asynccnt)
#define HAVE_ASYNC 1
#else
#define HAVE_ASYNC 0
#endif

static __device__ __forceinline__ unsigned short f2bf(float f) {
    unsigned int u = __builtin_bit_cast(unsigned int, f);
    u += 0x7FFFu + ((u >> 16) & 1u);              // round-to-nearest-even
    return (unsigned short)(u >> 16);
}

// Put a packed pair of bf16 (from one dword load) into vector elements i, i+1
static __device__ __forceinline__ void set2(v16bf& v, int i, unsigned int u) {
    v[i]     = __builtin_bit_cast(__bf16, (unsigned short)(u & 0xFFFFu));
    v[i + 1] = __builtin_bit_cast(__bf16, (unsigned short)(u >> 16));
}

// 16-byte global -> LDS copy. Async path uses GLOBAL_LOAD_ASYNC_TO_LDS_B128
// (ASYNCcnt); fallback is a synchronous load+store ordered by __syncthreads.
static __device__ __forceinline__ void copy16(const unsigned short* g, unsigned short* l) {
#if HAVE_ASYNC
    __builtin_amdgcn_global_load_async_to_lds_b128(
        (AS1 v4i*)(unsigned long long)g,
        (AS3 v4i*)(unsigned int)(unsigned long long)l,   // ISA: LDS addr = addr[31:0]
        0, 0);
#else
    *(uint4*)l = *(const uint4*)g;
#endif
}

static __device__ __forceinline__ void wait_async() {
#if HAVE_ASYNC
    __builtin_amdgcn_s_wait_asynccnt(0);
#endif
}

__global__ void __launch_bounds__(256)
cast_f32_bf16(const float* __restrict__ in, unsigned short* __restrict__ out, int n) {
    int i = blockIdx.x * blockDim.x + threadIdx.x;
    if (i < n) out[i] = f2bf(in[i]);
}

// ---------------------------------------------------------------------------
// QKV projection: qkv[r][e] = sum_d x[r][d] * w_qkv[e][d]
// One wave computes a 32x32 output tile (2x2 WMMA subtiles).
// Epilogue scatters into Q [B,H,T,HD], K [B,H,T,HD], Vt [B,H,HD,T] (bf16).
// ---------------------------------------------------------------------------
__global__ void __launch_bounds__(32)
gemm_qkv(const unsigned short* __restrict__ X,   // [NROW][DMOD] bf16
         const unsigned short* __restrict__ W,   // [QKVN][DMOD] bf16
         unsigned short* __restrict__ Qb,
         unsigned short* __restrict__ Kb,
         unsigned short* __restrict__ Vt) {
    const int lane = threadIdx.x;
    const int l15 = lane & 15, hgrp = lane >> 4;
    const int ntiles = QKVN / 32;                 // 72
    const int mbase = (blockIdx.x / ntiles) * 32;
    const int nbase = (blockIdx.x % ntiles) * 32;

    v8f acc[2][2] = {};
    for (int kk = 0; kk < DMOD; kk += 32) {
        v16bf a[2], bm[2];
#pragma unroll
        for (int mt = 0; mt < 2; ++mt) {
            const unsigned short* p = X + (mbase + mt * 16 + l15) * DMOD + kk;
#pragma unroll
            for (int v = 0; v < 8; ++v) {
                int k = hgrp * 16 + v * 2;                             // A layout
                set2(a[mt], 2 * v, *(const unsigned int*)(p + k));
            }
        }
#pragma unroll
        for (int nt = 0; nt < 2; ++nt) {
            const unsigned short* p = W + (nbase + nt * 16 + l15) * DMOD + kk;
#pragma unroll
            for (int v = 0; v < 8; ++v) {
                int k = hgrp * 16 + v * 2;                             // B layout
                set2(bm[nt], 2 * v, *(const unsigned int*)(p + k));
            }
        }
#pragma unroll
        for (int mt = 0; mt < 2; ++mt)
#pragma unroll
            for (int nt = 0; nt < 2; ++nt)
                acc[mt][nt] = __builtin_amdgcn_wmma_f32_16x16x32_bf16(
                    false, a[mt], false, bm[nt], (short)0, acc[mt][nt], false, false);
    }
    // Epilogue: scatter into head-major Q/K and transposed V
#pragma unroll
    for (int mt = 0; mt < 2; ++mt)
#pragma unroll
        for (int nt = 0; nt < 2; ++nt) {
            int n = nbase + nt * 16 + l15;       // 16-wide tile never crosses a 64/768 boundary
            int c = n / DMOD;
            int within = n % DMOD;
            int head = within / HDIM;
            int hd = within % HDIM;
#pragma unroll
            for (int v = 0; v < 8; ++v) {
                int m = mbase + mt * 16 + v + 8 * hgrp;               // C layout row
                int b = m / TLEN, t = m % TLEN;
                unsigned short val = f2bf(acc[mt][nt][v]);
                if (c == 0)      Qb[((b * NH + head) * TLEN + t) * HDIM + hd] = val;
                else if (c == 1) Kb[((b * NH + head) * TLEN + t) * HDIM + hd] = val;
                else             Vt[((b * NH + head) * HDIM + hd) * TLEN + t] = val;
            }
        }
}

// ---------------------------------------------------------------------------
// Flash attention, 4 waves per block sharing double-buffered K/V tiles in LDS.
// Block = (b, head, 64 query rows); wave w owns rows [qbase + 16w, +16).
// Per 32-key block: async copy of next K/V tile overlaps 8 WMMAs + softmax.
// ---------------------------------------------------------------------------
__global__ void __launch_bounds__(128)
flash_attn(const unsigned short* __restrict__ Qb,
           const unsigned short* __restrict__ Kb,
           const unsigned short* __restrict__ Vt,
           unsigned short* __restrict__ O) {      // [NROW][DMOD] bf16
    __shared__ __align__(16) unsigned short Kt [2][32][HDIM];   // [buf][key][hd]
    __shared__ __align__(16) unsigned short Vtl[2][HDIM][32];   // [buf][hd][key]
    __shared__ __align__(4)  unsigned short pbuf[4][16][34];    // per-wave P relayout

    const int tid  = threadIdx.x;
    const int lane = tid & 31;
    const int wid  = tid >> 5;
    const int l15 = lane & 15, hgrp = lane >> 4;

    const int qblocks = TLEN / 64;                // 64
    const int qt = blockIdx.x % qblocks;
    const int bh = blockIdx.x / qblocks;
    const int head = bh % NH, b = bh / NH;
    const int qbase = qt * 64 + wid * 16;

    const unsigned short* qptr = Qb + ((b * NH + head) * TLEN) * HDIM;
    const unsigned short* kptr = Kb + ((b * NH + head) * TLEN) * HDIM;
    const unsigned short* vptr = Vt + ((b * NH + head) * HDIM) * TLEN;

    // Cooperative tile copy: K tile 32x64 halves (4KB), V tile 64x32 halves (4KB).
    auto issue_copy = [&](int buf, int kv) {
        {   // K: 128 threads x 32B
            int row = tid >> 2, c = (tid & 3) * 16;
            const unsigned short* g = kptr + (kv + row) * HDIM + c;
            unsigned short* l = &Kt[buf][row][c];
            copy16(g, l);
            copy16(g + 8, l + 8);
        }
        {   // V: 128 threads x 32B
            int row = tid >> 1, c = (tid & 1) * 16;
            const unsigned short* g = vptr + row * TLEN + kv + c;
            unsigned short* l = &Vtl[buf][row][c];
            copy16(g, l);
            copy16(g + 8, l + 8);
        }
    };

    // Q fragments for both K=32 halves of HD=64
    v16bf qf[2];
#pragma unroll
    for (int kb = 0; kb < 2; ++kb) {
        const unsigned short* p = qptr + (qbase + l15) * HDIM + kb * 32;
#pragma unroll
        for (int v = 0; v < 8; ++v) {
            int k = hgrp * 16 + v * 2;
            set2(qf[kb], 2 * v, *(const unsigned int*)(p + k));
        }
    }

    v8f acc[4] = {};                              // O accumulators: 16 x 64
    float row_m[8], row_l[8];
#pragma unroll
    for (int v = 0; v < 8; ++v) { row_m[v] = -3.0e38f; row_l[v] = 0.0f; }
    const float scale = 0.125f;                   // 1/sqrt(64)

    issue_copy(0, 0);                             // prime the pipeline

    const int nkv = TLEN / 32;                    // 128 key blocks
    for (int kvb = 0; kvb < nkv; ++kvb) {
        const int buf = kvb & 1;
        wait_async();                             // this wave's copies into `buf` done
        __syncthreads();                          // all waves' copies done; buf^1 free
        if (kvb + 1 < nkv) issue_copy(buf ^ 1, (kvb + 1) * 32);

        // S = Q @ K^T from LDS (two 16x16 tiles; HD contraction in two K=32 steps)
        v8f s[2];
#pragma unroll
        for (int j = 0; j < 2; ++j) {
            v8f c = {};
#pragma unroll
            for (int kb = 0; kb < 2; ++kb) {
                v16bf bm;
                const unsigned short* p = &Kt[buf][j * 16 + l15][kb * 32 + hgrp * 16];
#pragma unroll
                for (int v = 0; v < 8; ++v)
                    set2(bm, 2 * v, *(const unsigned int*)(p + v * 2));
                c = __builtin_amdgcn_wmma_f32_16x16x32_bf16(
                    false, qf[kb], false, bm, (short)0, c, false, false);
            }
            s[j] = c;
        }
        // Online softmax (row reductions across the 16-lane half-groups)
#pragma unroll
        for (int v = 0; v < 8; ++v) {
            float s0 = s[0][v] * scale, s1 = s[1][v] * scale;
            float mx = fmaxf(s0, s1);
#pragma unroll
            for (int d = 1; d < 16; d <<= 1) mx = fmaxf(mx, __shfl_xor(mx, d, 32));
            float mn = fmaxf(row_m[v], mx);
            float al = __expf(row_m[v] - mn);
            float p0 = __expf(s0 - mn), p1 = __expf(s1 - mn);
            float sum = p0 + p1;
#pragma unroll
            for (int d = 1; d < 16; d <<= 1) sum += __shfl_xor(sum, d, 32);
            row_l[v] = row_l[v] * al + sum;
            row_m[v] = mn;
#pragma unroll
            for (int f = 0; f < 4; ++f) acc[f][v] *= al;
            int r = v + 8 * hgrp;                 // C-layout row this lane owns
            pbuf[wid][r][l15]      = f2bf(p0);
            pbuf[wid][r][16 + l15] = f2bf(p1);
        }
        // Wave-private relayout: same-wave DS ops execute in order (ISA 7.1);
        // only block compiler reordering.
        asm volatile("" ::: "memory");
        v16bf pa;
        {
            const unsigned short* p = &pbuf[wid][l15][0];
#pragma unroll
            for (int v = 0; v < 8; ++v) {
                int k = hgrp * 16 + v * 2;
                set2(pa, 2 * v, *(const unsigned int*)(p + k));
            }
        }
        asm volatile("" ::: "memory");
        // O += P @ V from LDS (V stored [hd][key])
#pragma unroll
        for (int f = 0; f < 4; ++f) {
            v16bf bm;
            const unsigned short* p = &Vtl[buf][f * 16 + l15][hgrp * 16];
#pragma unroll
            for (int v = 0; v < 8; ++v)
                set2(bm, 2 * v, *(const unsigned int*)(p + v * 2));
            acc[f] = __builtin_amdgcn_wmma_f32_16x16x32_bf16(
                false, pa, false, bm, (short)0, acc[f], false, false);
        }
    }
    // Normalize and store O in [B*T][D] bf16 for the projection GEMM
#pragma unroll
    for (int v = 0; v < 8; ++v) {
        int m = qbase + v + 8 * hgrp;
        float inv = 1.0f / row_l[v];
        int row = b * TLEN + m;
#pragma unroll
        for (int f = 0; f < 4; ++f)
            O[row * DMOD + head * HDIM + f * 16 + l15] = f2bf(acc[f][v] * inv);
    }
}

// ---------------------------------------------------------------------------
// Output projection: out[r][e] = sum_d O[r][d] * w_proj[e][d]   (fp32 out)
// ---------------------------------------------------------------------------
__global__ void __launch_bounds__(32)
gemm_proj(const unsigned short* __restrict__ A,   // [NROW][DMOD] bf16
          const unsigned short* __restrict__ W,   // [DMOD][DMOD] bf16
          float* __restrict__ out) {              // [NROW][DMOD] f32
    const int lane = threadIdx.x;
    const int l15 = lane & 15, hgrp = lane >> 4;
    const int ntiles = DMOD / 32;                 // 24
    const int mbase = (blockIdx.x / ntiles) * 32;
    const int nbase = (blockIdx.x % ntiles) * 32;

    v8f acc[2][2] = {};
    for (int kk = 0; kk < DMOD; kk += 32) {
        v16bf a[2], bm[2];
#pragma unroll
        for (int mt = 0; mt < 2; ++mt) {
            const unsigned short* p = A + (mbase + mt * 16 + l15) * DMOD + kk;
#pragma unroll
            for (int v = 0; v < 8; ++v) {
                int k = hgrp * 16 + v * 2;
                set2(a[mt], 2 * v, *(const unsigned int*)(p + k));
            }
        }
#pragma unroll
        for (int nt = 0; nt < 2; ++nt) {
            const unsigned short* p = W + (nbase + nt * 16 + l15) * DMOD + kk;
#pragma unroll
            for (int v = 0; v < 8; ++v)
                set2(bm[nt], 2 * v, *(const unsigned int*)(p + hgrp * 16 + v * 2));
        }
#pragma unroll
        for (int mt = 0; mt < 2; ++mt)
#pragma unroll
            for (int nt = 0; nt < 2; ++nt)
                acc[mt][nt] = __builtin_amdgcn_wmma_f32_16x16x32_bf16(
                    false, a[mt], false, bm[nt], (short)0, acc[mt][nt], false, false);
    }
#pragma unroll
    for (int mt = 0; mt < 2; ++mt)
#pragma unroll
        for (int nt = 0; nt < 2; ++nt) {
            int n = nbase + nt * 16 + l15;
#pragma unroll
            for (int v = 0; v < 8; ++v) {
                int m = mbase + mt * 16 + v + 8 * hgrp;
                out[m * DMOD + n] = acc[mt][nt][v];
            }
        }
}

// ---------------------------------------------------------------------------
extern "C" void kernel_launch(void* const* d_in, const int* in_sizes, int n_in,
                              void* d_out, int out_size, void* d_ws, size_t ws_size,
                              hipStream_t stream) {
    const float* x      = (const float*)d_in[0];   // [B,T,D]
    const float* w_qkv  = (const float*)d_in[1];   // [3D,D]
    const float* w_proj = (const float*)d_in[2];   // [D,D]
    float* out = (float*)d_out;                    // [B,T,D]

    char* ws = (char*)d_ws;
    size_t off = 0;
    auto carve = [&](size_t bytes) -> void* {
        void* p = ws + off;
        off += (bytes + 255) & ~(size_t)255;
        return p;
    };
    unsigned short* xb     = (unsigned short*)carve((size_t)NROW * DMOD * 2);
    unsigned short* wqkvb  = (unsigned short*)carve((size_t)QKVN * DMOD * 2);
    unsigned short* wprojb = (unsigned short*)carve((size_t)DMOD * DMOD * 2);
    unsigned short* Qb     = (unsigned short*)carve((size_t)BSZ * NH * TLEN * HDIM * 2);
    unsigned short* Kb     = (unsigned short*)carve((size_t)BSZ * NH * TLEN * HDIM * 2);
    unsigned short* Vt     = (unsigned short*)carve((size_t)BSZ * NH * TLEN * HDIM * 2);
    unsigned short* Ob     = (unsigned short*)carve((size_t)NROW * DMOD * 2);

    int nx = NROW * DMOD, nw1 = QKVN * DMOD, nw2 = DMOD * DMOD;
    cast_f32_bf16<<<(nx  + 255) / 256, 256, 0, stream>>>(x,      xb,     nx);
    cast_f32_bf16<<<(nw1 + 255) / 256, 256, 0, stream>>>(w_qkv,  wqkvb,  nw1);
    cast_f32_bf16<<<(nw2 + 255) / 256, 256, 0, stream>>>(w_proj, wprojb, nw2);

    gemm_qkv<<<(NROW / 32) * (QKVN / 32), 32, 0, stream>>>(xb, wqkvb, Qb, Kb, Vt);
    flash_attn<<<BSZ * NH * (TLEN / 64), 128, 0, stream>>>(Qb, Kb, Vt, Ob);
    gemm_proj<<<(NROW / 32) * (DMOD / 32), 32, 0, stream>>>(Ob, wprojb, out);
}
